// HardLSTM_56083682951501
// MI455X (gfx1250) — matrix operand order repaired and checked by reading, not compiled, not run on hardware
//
#include <hip/hip_runtime.h>
#include <hip/hip_bf16.h>
#include <stdint.h>

// ---------------------------------------------------------------------------
// HardLSTM (2-layer bidirectional) for gfx1250 (MI455X, wave32, WMMA).
//   T=512, B=32, IN=320, H=1024, gates 4H=4096.
//   * f16 data / f32 accumulate via v_wmma_f32_16x16x32_f16
//   * input projections: batched GEMM, B-tile staged in LDS with
//     global_load_async_to_lds_b128 (double buffered, ASYNCcnt-tracked)
//   * recurrence: ONE persistent kernel per layer (fwd+bwd fused), timesteps
//     separated by a device-scope atomic grid barrier (32 workgroups total ->
//     co-residency guaranteed). Cell state c lives in registers all 512 steps.
//   * weights stay L2-resident; single-use pre-activation / output traffic is
//     tagged non-temporal so it cannot evict them.
// Workspace requirement: ~420 MB.
// ---------------------------------------------------------------------------

#define T_ 512
#define B_ 32
#define IN_ 320
#define H_ 1024
#define G4H_ 4096

typedef _Float16 v16h __attribute__((ext_vector_type(16)));
typedef _Float16 v8h  __attribute__((ext_vector_type(8)));
typedef float    v8f  __attribute__((ext_vector_type(8)));

struct alignas(128) GBar { unsigned count; unsigned gen; };

__device__ __forceinline__ float hsig_(float x) {
    return __builtin_fminf(__builtin_fmaxf(0.2f * x + 0.5f, 0.0f), 1.0f);
}
__device__ __forceinline__ float htanh_(float x) {
    return __builtin_fminf(__builtin_fmaxf(x, -1.0f), 1.0f);
}

// Generation-counting grid barrier (per direction, nb workgroups).
__device__ __forceinline__ void gbar_sync(GBar* b, unsigned nb) {
    __threadfence();          // release this block's stores to device scope
    __syncthreads();
    if (threadIdx.x == 0) {
        unsigned g = __hip_atomic_load(&b->gen, __ATOMIC_RELAXED,
                                       __HIP_MEMORY_SCOPE_AGENT);
        unsigned a = __hip_atomic_fetch_add(&b->count, 1u, __ATOMIC_ACQ_REL,
                                            __HIP_MEMORY_SCOPE_AGENT);
        if (a == nb - 1) {
            __hip_atomic_store(&b->count, 0u, __ATOMIC_RELAXED,
                               __HIP_MEMORY_SCOPE_AGENT);
            __hip_atomic_store(&b->gen, g + 1u, __ATOMIC_RELEASE,
                               __HIP_MEMORY_SCOPE_AGENT);
        } else {
            while (__hip_atomic_load(&b->gen, __ATOMIC_ACQUIRE,
                                     __HIP_MEMORY_SCOPE_AGENT) == g)
                __builtin_amdgcn_s_sleep(2);
        }
    }
    __syncthreads();
    __threadfence();          // acquire: no stale lines for next step's reads
}

// A-fragment (16x32 f16, M x K), row-major source with leading dim ld.
// lanes 0-15 hold row M=lane, K = k0..k0+15;
//             lanes 16-31 same rows, K shifted by +16.
__device__ __forceinline__ v16h load_a_frag(const _Float16* __restrict__ p,
                                            int ld, int m0, int k0) {
    const int lane = threadIdx.x & 31;
    const int m   = m0 + (lane & 15);
    const int klo = k0 + ((lane >> 4) << 4);
    const _Float16* q = p + (size_t)m * ld + klo;
    v8h lo = *(const v8h*)q;
    v8h hi = *(const v8h*)(q + 8);
    v16h a;
#pragma unroll
    for (int i = 0; i < 8; ++i) { a[i] = lo[i]; a[i + 8] = hi[i]; }
    return a;
}

// B-fragment (32x16 f16, K x N) from W row-major [N_rows x K]: B(k,n)=W[n][k].
// lanes 0-15: N=lane, K=k0..k0+15; lanes 16-31: K += 16.
__device__ __forceinline__ v16h load_b_frag(const _Float16* __restrict__ w,
                                            int ld, int n0, int k0) {
    const int lane = threadIdx.x & 31;
    const int n  = n0 + (lane & 15);
    const int kk = k0 + ((lane >> 4) << 4);
    return *(const v16h*)(w + (size_t)n * ld + kk);
}

// Same B-fragment but from an LDS tile of 64 rows x 32 halfs (row-major).
__device__ __forceinline__ v16h load_b_frag_lds(const _Float16* lb, int nloc) {
    const int lane = threadIdx.x & 31;
    const int n  = nloc + (lane & 15);
    const int kk = (lane >> 4) << 4;
    return *(const v16h*)(lb + n * 32 + kk);
}

// --------------------------- conversion kernel -----------------------------
__global__ __launch_bounds__(256) void hardlstm_cvt_f32_f16(
    const float* __restrict__ s, _Float16* __restrict__ d, int n) {
    int i = blockIdx.x * blockDim.x + threadIdx.x;
    int stride = gridDim.x * blockDim.x;
    for (; i < n; i += stride) d[i] = (_Float16)s[i];
}

// --------------------- batched input-projection GEMM -----------------------
// out[m,n] = sum_k A[m,k]*W[n,k] + bih[n] + bhh[n]; N fixed = 4096, f16 out.
// Block = 8 waves covering a 128(M) x 64(N) tile: all waves share the B strip,
// staged through LDS with async global->LDS copies (double buffer).
// NOTE: M must be a multiple of 128 (true here: 16384) - no early returns
// allowed because every wave participates in barriers.
__global__ __launch_bounds__(256) void hardlstm_pre_gemm(
    const _Float16* __restrict__ A, const _Float16* __restrict__ W,
    const float* __restrict__ bih, const float* __restrict__ bhh,
    _Float16* __restrict__ out, int M, int K) {
    __shared__ _Float16 ldsW[2][64 * 32];  // two 4KB buffers: 64 rows x 32 halfs

    const int tid  = threadIdx.x;
    const int lane = tid & 31;
    const int wv   = tid >> 5;                    // 0..7
    const int mblock = blockIdx.x >> 6;           // 64 strips in N
    const int strip  = blockIdx.x & 63;
    const int m0  = (mblock << 7) + (wv << 4);
    const int n0s = strip << 6;
    const int nl  = lane & 15;
    const int hb8 = (lane >> 4) << 3;

    // async staging: thread tid copies one 16B segment per chunk
    const uint32_t lds_base = (uint32_t)(uint64_t)(&ldsW[0][0]);
    const int cp_row = tid >> 2;                  // 0..63
    const int cp_seg = (tid & 3) << 3;            // 0,8,16,24 (halfs)
    const _Float16* wsrc = W + (size_t)(n0s + cp_row) * K + cp_seg;
    auto stage = [&](int k0, int buf) {
        uint32_t dst = lds_base +
            (uint32_t)(buf * (64 * 32) + cp_row * 32 + cp_seg) * 2u;
        const void* src = (const void*)(wsrc + k0);
        asm volatile("global_load_async_to_lds_b128 %0, %1, off"
                     :: "v"(dst), "v"(src) : "memory");
    };

    v8f acc[4];
#pragma unroll
    for (int j = 0; j < 4; ++j) {
        const int n = n0s + (j << 4) + nl;
        const float bv = bih[n] + bhh[n];
#pragma unroll
        for (int e = 0; e < 8; ++e) acc[j][e] = bv;
    }

    stage(0, 0);
    v16h a_cur = load_a_frag(A, K, m0, 0);

    const int nk = K >> 5;
#pragma unroll 2
    for (int i = 0; i < nk; ++i) {
        asm volatile("s_wait_asynccnt 0x0" ::: "memory");
        __syncthreads();                          // LDS buffer (i&1) ready
        v16h a_nxt = a_cur;
        if (i + 1 < nk) {
            stage((i + 1) << 5, (i + 1) & 1);     // overlap copy with compute
            a_nxt = load_a_frag(A, K, m0, (i + 1) << 5);
        }
        const _Float16* lb = &ldsW[i & 1][0];
        v16h b[4];
#pragma unroll
        for (int j = 0; j < 4; ++j) b[j] = load_b_frag_lds(lb, j << 4);
#pragma unroll
        for (int j = 0; j < 4; ++j)
            acc[j] = __builtin_amdgcn_wmma_f32_16x16x32_f16(
                false, a_cur, false, b[j], (short)0, acc[j], false, false);
        a_cur = a_nxt;
    }

    // single-use pre-activations: non-temporal stores (protect L2 weights)
#pragma unroll
    for (int j = 0; j < 4; ++j) {
        const int n = n0s + (j << 4) + nl;
#pragma unroll
        for (int e = 0; e < 8; ++e)
            __builtin_nontemporal_store(
                (_Float16)acc[j][e], out + (size_t)(m0 + e + hb8) * G4H_ + n);
    }
}

// ----------------------- persistent LSTM scan ------------------------------
// One launch per layer. grid = (16, 2): 16 workgroups per direction, each
// wave owns a 16(M) x 16(N-hidden) chunk for all 512 steps; c stays in VGPRs.
// Timesteps separated by a per-direction atomic grid barrier.
__global__ __launch_bounds__(256) void hardlstm_scan(
    const _Float16* __restrict__ pre16,  // [2][T*B][4096]
    const _Float16* __restrict__ whh,    // [2][4096][1024]
    _Float16* __restrict__ hping,        // [2][B][H] (f16, ping)
    _Float16* __restrict__ hpong,        // [2][B][H] (f16, pong)
    const float* __restrict__ c0slice,   // [2][B][H] initial cell state
    _Float16* __restrict__ y16,          // [T*B][2H] or null (layer0 output)
    float* __restrict__ y32,             // [T*B][2H] or null (final output)
    float* __restrict__ outHn,           // [2][B][H] final h -> d_out
    float* __restrict__ outCn,           // [2][B][H] final c -> d_out
    GBar* __restrict__ bar) {            // [2]
    const int dir  = blockIdx.y;
    const int lane = threadIdx.x & 31;
    const int wid  = blockIdx.x * (blockDim.x >> 5) + (threadIdx.x >> 5);
    const int m0   = (wid >> 6) << 4;  // 0 or 16 (batch tile)
    const int n0   = (wid & 63) << 4;  // hidden strip
    const int nl   = lane & 15;
    const int hb8  = (lane >> 4) << 3;

    const _Float16* preDir = pre16 + (size_t)dir * T_ * B_ * G4H_;
    const _Float16* wD     = whh + (size_t)dir * G4H_ * H_;
    const size_t dirOff    = (size_t)dir * B_ * H_;
    _Float16* hp0 = hping + dirOff;
    _Float16* hp1 = hpong + dirOff;
    GBar* mybar = bar + dir;

    // cell state lives in registers for the whole scan
    v8f creg;
#pragma unroll
    for (int e = 0; e < 8; ++e)
        creg[e] = c0slice[dirOff + (size_t)(m0 + e + hb8) * H_ + n0 + nl];

    for (int t = 0; t < T_; ++t) {
        const int tm = dir ? (T_ - 1 - t) : t;
        const _Float16* hs = (t & 1) ? hp1 : hp0;
        _Float16*       hd = (t & 1) ? hp0 : hp1;
        const _Float16* preD = preDir + (size_t)tm * B_ * G4H_;

        // issue pre-activation loads early (non-temporal, single use);
        // consumed only in the epilogue -> hidden behind the K loop
        float pf[4][8];
#pragma unroll
        for (int g = 0; g < 4; ++g)
#pragma unroll
            for (int e = 0; e < 8; ++e)
                pf[g][e] = (float)__builtin_nontemporal_load(
                    preD + (size_t)(m0 + e + hb8) * G4H_ + g * H_ + n0 + nl);

        v8f acc[4];
#pragma unroll
        for (int g = 0; g < 4; ++g)
#pragma unroll
            for (int e = 0; e < 8; ++e) acc[g][e] = 0.0f;

        // register double-buffered K loop: loads for k+32 overlap WMMAs for k
        v16h a_cur = load_a_frag(hs, H_, m0, 0);
        v16h b_cur[4];
#pragma unroll
        for (int g = 0; g < 4; ++g)
            b_cur[g] = load_b_frag(wD, H_, g * H_ + n0, 0);

#pragma unroll 2
        for (int k0 = 0; k0 < H_; k0 += 32) {
            v16h a_nxt = a_cur;
            v16h b_nxt[4];
            const int k1 = k0 + 32;
            if (k1 < H_) {
                a_nxt = load_a_frag(hs, H_, m0, k1);
#pragma unroll
                for (int g = 0; g < 4; ++g)
                    b_nxt[g] = load_b_frag(wD, H_, g * H_ + n0, k1);
            } else {
#pragma unroll
                for (int g = 0; g < 4; ++g) b_nxt[g] = b_cur[g];
            }
#pragma unroll
            for (int g = 0; g < 4; ++g)
                acc[g] = __builtin_amdgcn_wmma_f32_16x16x32_f16(
                    false, a_cur, false, b_cur[g], (short)0, acc[g], false, false);
            a_cur = a_nxt;
#pragma unroll
            for (int g = 0; g < 4; ++g) b_cur[g] = b_nxt[g];
        }

        // fused hard-LSTM cell update
#pragma unroll
        for (int e = 0; e < 8; ++e) {
            const int m = m0 + e + hb8;
            const int n = n0 + nl;
            const size_t idx = (size_t)m * H_ + n;
            const float iv = hsig_(acc[0][e] + pf[0][e]);
            const float fv = hsig_(acc[1][e] + pf[1][e]);
            const float gv = htanh_(acc[2][e] + pf[2][e]);
            const float ov = hsig_(acc[3][e] + pf[3][e]);
            const float cn = fv * creg[e] + iv * gv;
            creg[e] = cn;
            const float hn = ov * htanh_(cn);
            hd[idx] = (_Float16)hn;
            const size_t yidx =
                ((size_t)tm * B_ + m) * (2 * H_) + (size_t)dir * H_ + n;
            if (y16) __builtin_nontemporal_store((_Float16)hn, y16 + yidx);
            if (y32) __builtin_nontemporal_store(hn, y32 + yidx);
            if (t == T_ - 1) {
                outHn[dirOff + idx] = hn;
                outCn[dirOff + idx] = cn;
            }
        }

        if (t + 1 < T_) gbar_sync(mybar, gridDim.x);
    }
}

// ---------------------------------------------------------------------------
extern "C" void kernel_launch(void* const* d_in, const int* in_sizes, int n_in,
                              void* d_out, int out_size, void* d_ws, size_t ws_size,
                              hipStream_t stream) {
    (void)in_sizes; (void)n_in; (void)out_size; (void)ws_size;

    const float* x  = (const float*)d_in[0];
    const float* h0 = (const float*)d_in[1];
    const float* c0 = (const float*)d_in[2];
    // weights: [3..6]=l0f, [7..10]=l0b, [11..14]=l1f, [15..18]=l1b
    // within each: w_ih, w_hh, b_ih, b_hh
    float* outF = (float*)d_out;

    // ---- carve workspace -------------------------------------------------
    size_t off = 0;
    auto carve = [&](size_t bytes) -> char* {
        char* p = (char*)d_ws + off;
        off += (bytes + 255) & ~(size_t)255;
        return p;
    };
    _Float16* x16   = (_Float16*)carve((size_t)T_ * B_ * IN_ * 2);
    _Float16* y016  = (_Float16*)carve((size_t)T_ * B_ * 2 * H_ * 2);
    _Float16* wih0  = (_Float16*)carve((size_t)2 * G4H_ * IN_ * 2);       // [2][4096][320]
    _Float16* wih1  = (_Float16*)carve((size_t)2 * G4H_ * 2 * H_ * 2);    // [2][4096][2048]
    _Float16* whh0  = (_Float16*)carve((size_t)2 * G4H_ * H_ * 2);        // [2][4096][1024]
    _Float16* whh1  = (_Float16*)carve((size_t)2 * G4H_ * H_ * 2);
    _Float16* pre16 = (_Float16*)carve((size_t)2 * T_ * B_ * G4H_ * 2);   // [2][T*B][4096]
    _Float16* hping = (_Float16*)carve((size_t)2 * B_ * H_ * 2);          // [2][B][H]
    _Float16* hpong = (_Float16*)carve((size_t)2 * B_ * H_ * 2);
    GBar*     bar   = (GBar*)carve(sizeof(GBar) * 2);

    auto cvt = [&](const void* s, _Float16* d, int n) {
        int blocks = (n + 255) / 256;
        if (blocks > 16384) blocks = 16384;
        hipLaunchKernelGGL(hardlstm_cvt_f32_f16, dim3(blocks), dim3(256), 0, stream,
                           (const float*)s, d, n);
    };

    // barrier state must start from a known value every call
    hipMemsetAsync(bar, 0, sizeof(GBar) * 2, stream);

    // ---- one-time conversions -------------------------------------------
    cvt(x, x16, T_ * B_ * IN_);
    cvt(d_in[3],  wih0,                      G4H_ * IN_);
    cvt(d_in[7],  wih0 + (size_t)G4H_ * IN_, G4H_ * IN_);
    cvt(d_in[4],  whh0,                      G4H_ * H_);
    cvt(d_in[8],  whh0 + (size_t)G4H_ * H_,  G4H_ * H_);
    cvt(d_in[11], wih1,                          G4H_ * 2 * H_);
    cvt(d_in[15], wih1 + (size_t)G4H_ * 2 * H_,  G4H_ * 2 * H_);
    cvt(d_in[12], whh1,                      G4H_ * H_);
    cvt(d_in[16], whh1 + (size_t)G4H_ * H_,  G4H_ * H_);

    const int M = T_ * B_;                       // 16384 rows
    const int gemm_blocks = (M / 128) * 64;      // 128-row M blocks x 64 N strips
    const size_t yElems = (size_t)T_ * B_ * 2 * H_;

    for (int layer = 0; layer < 2; ++layer) {
        const int K = (layer == 0) ? IN_ : 2 * H_;
        const _Float16* Ain = (layer == 0) ? x16 : y016;
        const _Float16* wih = (layer == 0) ? wih0 : wih1;
        const _Float16* whh = (layer == 0) ? whh0 : whh1;
        const int bihIdx[2] = { layer == 0 ? 5 : 13, layer == 0 ? 9 : 17 };
        const int bhhIdx[2] = { layer == 0 ? 6 : 14, layer == 0 ? 10 : 18 };

        // h_init (f16) for this layer's two directions
        cvt(h0 + (size_t)layer * 2 * B_ * H_, hping, 2 * B_ * H_);

        // batched input projections (both directions)
        for (int dir = 0; dir < 2; ++dir) {
            hipLaunchKernelGGL(hardlstm_pre_gemm, dim3(gemm_blocks), dim3(256), 0, stream,
                               Ain, wih + (size_t)dir * G4H_ * K,
                               (const float*)d_in[bihIdx[dir]],
                               (const float*)d_in[bhhIdx[dir]],
                               pre16 + (size_t)dir * T_ * B_ * G4H_, M, K);
        }

        // one persistent scan kernel for all 512 steps (fwd+bwd fused)
        _Float16* y16 = (layer == 0) ? y016 : nullptr;
        float*    y32 = (layer == 1) ? outF : nullptr;
        float* outHn = outF + yElems + (size_t)layer * 2 * B_ * H_;
        float* outCn = outF + yElems + (size_t)4 * B_ * H_ + (size_t)layer * 2 * B_ * H_;
        hipLaunchKernelGGL(hardlstm_scan, dim3(16, 2), dim3(256), 0, stream,
                           pre16, whh, hping, hpong,
                           c0 + (size_t)layer * 2 * B_ * H_,
                           y16, y32, outHn, outCn, bar);
    }
}
